// QuantLinear_18013092839920
// MI455X (gfx1250) — matrix operand order repaired and checked by reading, not compiled, not run on hardware
//
#include <hip/hip_runtime.h>
#include <stdint.h>

typedef __attribute__((ext_vector_type(16))) _Float16 v16h;
typedef __attribute__((ext_vector_type(8)))  _Float16 v8h;
typedef __attribute__((ext_vector_type(4)))  _Float16 h4;
typedef __attribute__((ext_vector_type(8)))  float    v8f;

#define MDIM 4096
#define NDIM 4096
#define KDIM 4096
#define GROUP 128
#define NGROUPS (KDIM / GROUP)   // 32

#define BM 128
#define BN 128
#define BK 64
#define LDSP 72                  // padded row stride in halves (144B = 36 dwords, conflict-free)
#define KTILES (KDIM / BK)       // 64

// ---------------- CDNA5 async global->LDS helpers (ASYNCcnt-tracked) --------
__device__ __forceinline__ void async_b128_to_lds(unsigned lds_addr, const void* gaddr) {
    // VDST = per-lane LDS byte address VGPR, VADDR = 64-bit global address.
    asm volatile("global_load_async_to_lds_b128 %0, %1, off"
                 :: "v"(lds_addr), "v"(gaddr) : "memory");
}
__device__ __forceinline__ void wait_async0() {
    asm volatile("s_wait_asynccnt 0x0" ::: "memory");
}
// Generic LDS pointers carry the LDS byte offset in their low 32 bits.
__device__ __forceinline__ unsigned lds_off(const void* p) {
    return (unsigned)(uintptr_t)p;
}

// Reference fp4 quant-dequant (E2M1 with mantissa-clamping decode) folded with
// the scale: fp4_qd(w/s)*s with no division, as an exact comparison network
// on scaled thresholds. Ties at 0.25/1.25/2.5/5 follow round-half-even (down).
__device__ __forceinline__ _Float16 qd1(float w, float s) {
    float a = fabsf(w);
    float r;
    if      (a <= 0.25f * s) r = 0.0f;
    else if (a <  s)         r = 0.5f * s;   // subnormal region, mantissa clamped
    else if (a <= 1.25f * s) r = s;
    else if (a <  2.0f * s)  r = 1.5f * s;
    else if (a <= 2.5f * s)  r = 2.0f * s;
    else if (a <  4.0f * s)  r = 3.0f * s;
    else if (a <= 5.0f * s)  r = 4.0f * s;
    else                     r = 6.0f * s;
    return (_Float16)copysignf(r, w);
}

// One block per (row, group): amax over 128 weights -> scale = amax/6 (1 if 0).
__global__ void k_scales(const float* __restrict__ W, float* __restrict__ S) {
    const int g = blockIdx.x;
    const int t = threadIdx.x;           // 0..127
    __shared__ float red[128];
    red[t] = fabsf(W[(size_t)g * GROUP + t]);
    __syncthreads();
    for (int s = 64; s > 0; s >>= 1) {
        if (t < s) red[t] = fmaxf(red[t], red[t + s]);
        __syncthreads();
    }
    if (t == 0) {
        float amax = red[0];
        S[g] = (amax == 0.0f) ? 1.0f : amax / 6.0f;
    }
}

// w (f32) -> quant-dequant -> f16 w_deq. One float4 per thread.
__global__ void k_dequant(const float* __restrict__ W, const float* __restrict__ S,
                          _Float16* __restrict__ Wq) {
    const int e4 = blockIdx.x * blockDim.x + threadIdx.x;
    const int n  = e4 >> 10;                    // 1024 float4 per row
    const int c4 = e4 & 1023;
    const float s = S[n * NGROUPS + (c4 >> 5)];
    const float4 w4 = *(const float4*)&W[(size_t)e4 * 4];
    h4 h;
    h[0] = qd1(w4.x, s); h[1] = qd1(w4.y, s);
    h[2] = qd1(w4.z, s); h[3] = qd1(w4.w, s);
    *(h4*)&Wq[(size_t)e4 * 4] = h;
}

// x (f32) -> f16, one float4 per thread (memory-bound).
__global__ void k_xhalf(const float* __restrict__ X, _Float16* __restrict__ Xh) {
    const int e4 = blockIdx.x * blockDim.x + threadIdx.x;
    const float4 x4 = *(const float4*)&X[(size_t)e4 * 4];
    h4 h;
    h[0] = (_Float16)x4.x; h[1] = (_Float16)x4.y;
    h[2] = (_Float16)x4.z; h[3] = (_Float16)x4.w;
    *(h4*)&Xh[(size_t)e4 * 4] = h;
}

__global__ __launch_bounds__(256) void k_gemm(const _Float16* __restrict__ Xh,
                                              const _Float16* __restrict__ Wq,
                                              const float* __restrict__ Bias,
                                              float* __restrict__ Out) {
    __shared__ _Float16 As[2][BM * LDSP];
    __shared__ _Float16 Bs[2][BN * LDSP];

    const int tid  = threadIdx.x;
    const int lane = tid & 31;
    const int wave = tid >> 5;           // 0..7
    const int wm   = wave >> 2;          // 0..1  (64 rows each)
    const int wn   = wave & 3;           // 0..3  (32 cols each)
    const int m0   = blockIdx.y * BM;
    const int n0   = blockIdx.x * BN;
    const int lh   = lane & 15;
    const int sel  = lane >> 4;          // half-wave selector

    v8f acc[4][2];
#pragma unroll
    for (int i = 0; i < 4; ++i)
#pragma unroll
        for (int j = 0; j < 2; ++j)
            acc[i][j] = (v8f){0.f, 0.f, 0.f, 0.f, 0.f, 0.f, 0.f, 0.f};

    // Tile = 128 rows x 64 halves = 1024 b128 chunks; 4 per thread.
    // chunk c -> row r = c>>3, 8-half unit cv = c&7.
    // Precompute per-thread chunk geometry.
    int rr[4], cc[4];
#pragma unroll
    for (int t = 0; t < 4; ++t) {
        const int c = tid + t * 256;
        rr[t] = c >> 3;
        cc[t] = (c & 7) * 8;             // half offset within the 64-wide row
    }

    // ---- prologue: async-fill buffer 0 with tile 0 ----
#pragma unroll
    for (int t = 0; t < 4; ++t) {
        async_b128_to_lds(lds_off(&As[0][rr[t] * LDSP + cc[t]]),
                          Xh + (size_t)(m0 + rr[t]) * KDIM + cc[t]);
        async_b128_to_lds(lds_off(&Bs[0][rr[t] * LDSP + cc[t]]),
                          Wq + (size_t)(n0 + rr[t]) * KDIM + cc[t]);
    }
    wait_async0();
    __syncthreads();

    for (int kt = 0; kt < KTILES; ++kt) {
        const int cur = kt & 1;
        const int nxt = cur ^ 1;

        // ---- async-fill the other buffer with tile kt+1 (overlaps WMMA) ----
        if (kt + 1 < KTILES) {
            const int k0 = (kt + 1) * BK;
#pragma unroll
            for (int t = 0; t < 4; ++t) {
                async_b128_to_lds(lds_off(&As[nxt][rr[t] * LDSP + cc[t]]),
                                  Xh + (size_t)(m0 + rr[t]) * KDIM + k0 + cc[t]);
                async_b128_to_lds(lds_off(&Bs[nxt][rr[t] * LDSP + cc[t]]),
                                  Wq + (size_t)(n0 + rr[t]) * KDIM + k0 + cc[t]);
            }
        }

        // ---- two K=32 WMMA steps over the 64-deep tile ----
#pragma unroll
        for (int ks = 0; ks < 2; ++ks) {
            const int ko = ks * 32;
            // A 16x32: lane<16 holds K 0-7 & 16-23 of row M=lane&15; lane>=16: K 8-15 & 24-31.
            v16h afr[4], bfr[2];
#pragma unroll
            for (int i = 0; i < 4; ++i) {
                const int m = wm * 64 + i * 16 + lh;
                const v8h lo = *(const v8h*)&As[cur][m * LDSP + ko + sel * 8];
                const v8h hi = *(const v8h*)&As[cur][m * LDSP + ko + sel * 8 + 16];
                v16h a;
#pragma unroll
                for (int q = 0; q < 8; ++q) { a[q] = lo[q]; a[q + 8] = hi[q]; }
                afr[i] = a;
            }
            // B 32x16: lane<16 -> N=lane&15, K 0-7 & 16-23; lane>=16 -> K 8-15 & 24-31.
#pragma unroll
            for (int j = 0; j < 2; ++j) {
                const int n = wn * 32 + j * 16 + lh;
                const v8h lo = *(const v8h*)&Bs[cur][n * LDSP + ko + sel * 8];
                const v8h hi = *(const v8h*)&Bs[cur][n * LDSP + ko + sel * 8 + 16];
                v16h b;
#pragma unroll
                for (int q = 0; q < 8; ++q) { b[q] = lo[q]; b[q + 8] = hi[q]; }
                bfr[j] = b;
            }
#pragma unroll
            for (int i = 0; i < 4; ++i)
#pragma unroll
                for (int j = 0; j < 2; ++j)
                    acc[i][j] = __builtin_amdgcn_wmma_f32_16x16x32_f16(
                        /*neg_a=*/false, afr[i], /*neg_b=*/false, bfr[j],
                        /*c_mod=*/(short)0, acc[i][j],
                        /*reuse_a=*/false, /*reuse_b=*/false);
        }

        // Own async fills must land before everyone crosses the barrier.
        if (kt + 1 < KTILES) wait_async0();
        __syncthreads();
    }

    // ---- epilogue: bias add + store (C layout: N=lane&15, M=vgpr + 8*sel) ----
#pragma unroll
    for (int j = 0; j < 2; ++j) {
        const int n  = n0 + wn * 32 + j * 16 + lh;
        const float bv = Bias[n];
#pragma unroll
        for (int i = 0; i < 4; ++i) {
            const int mbase = m0 + wm * 64 + i * 16 + sel * 8;
#pragma unroll
            for (int r = 0; r < 8; ++r)
                Out[(size_t)(mbase + r) * NDIM + n] = acc[i][j][r] + bv;
        }
    }
}

extern "C" void kernel_launch(void* const* d_in, const int* in_sizes, int n_in,
                              void* d_out, int out_size, void* d_ws, size_t ws_size,
                              hipStream_t stream) {
    const float* x = (const float*)d_in[0];   // [4096,4096]
    const float* w = (const float*)d_in[1];   // [4096,4096]
    const float* b = (const float*)d_in[2];   // [4096]
    float* out = (float*)d_out;               // [4096,4096]

    // Workspace: [0,512KB) scales f32; +512KB f16 w_deq (32MB); +32.5MB f16 x (32MB).
    float*    scales = (float*)d_ws;
    _Float16* wq = (_Float16*)((char*)d_ws + (size_t)NDIM * NGROUPS * sizeof(float));
    _Float16* xh = wq + (size_t)NDIM * KDIM;

    k_scales<<<NDIM * NGROUPS, 128, 0, stream>>>(w, scales);
    k_dequant<<<(NDIM * (KDIM / 4)) / 256, 256, 0, stream>>>(w, scales, wq);
    k_xhalf<<<(MDIM * (KDIM / 4)) / 256, 256, 0, stream>>>(x, xh);

    dim3 grid(NDIM / BN, MDIM / BM);
    k_gemm<<<grid, 256, 0, stream>>>(xh, wq, b, out);
}
